// LinearChainCRF_1726576857492
// MI455X (gfx1250) — matrix operand order repaired and checked by reading, not compiled, not run on hardware
//
#include <hip/hip_runtime.h>

// MI455X / gfx1250, wave32. CRF forward scan:
//  - Kernel 1 (crf_path_score): supervised channel collapses to exact path
//    score (IMPOSSIBLE=-1e4 guarantees exp underflow -> logsumexp == max in
//    f32). Chunked 32 steps/iter per wave, 64 waves. Writes z0 to d_ws.
//  - Kernel 2 (crf_unsup_scan): unsupervised forward scan, 16 chains/wave,
//    carried in BASE-2 log domain so exp/log are raw v_exp_f32/v_log_f32.
//    Per step: row-max (shfl), p = exp2(alpha2-m2) -> LDS transpose -> A,
//    8x v_wmma_f32_16x16x32_f16 against resident E = 2^(trans*log2e) B-tiles,
//    alpha2' = (m2 + em2) + log2(s). Emissions software-pipelined one step
//    ahead; `last`-latch loop split at minLen-1 so the early half of the scan
//    carries no select overhead.

typedef __attribute__((ext_vector_type(16))) _Float16 v16h;
typedef __attribute__((ext_vector_type(8)))  _Float16 v8h;
typedef __attribute__((ext_vector_type(8)))  float    v8f;

#define T_LEN 4096
#define NTAG  64
#define IMPOSSIBLE_F (-10000.0f)
#define L2E 1.44269504088896340736f   // log2(e)
#define LN2 0.69314718055994530942f   // ln(2)

__device__ __forceinline__ int byteSum(unsigned x) {
  x = (x & 0x00FF00FFu) + ((x >> 8) & 0x00FF00FFu);
  return (int)((x & 0xFFFFu) + (x >> 16));
}

__device__ __forceinline__ int tagFromDword(unsigned w, int off) {
  return w ? (off + ((__ffs(w) - 1) >> 3)) : -1;
}

// ---------------------------------------------------------------------------
// Supervised channel: exact path score, z0 -> ws[b].
// One wave per batch; 32 time-steps per iteration (tag extraction is
// data-independent of the accumulator -> latency fully pipelined).
// ---------------------------------------------------------------------------
__global__ __launch_bounds__(32, 1)
void crf_path_score(const float* __restrict__ em,
                    const unsigned char* __restrict__ mask,
                    const unsigned char* __restrict__ target,
                    const float* __restrict__ trans,
                    const float* __restrict__ startt,
                    const float* __restrict__ endt,
                    const unsigned char* __restrict__ ftrans,
                    const unsigned char* __restrict__ fstart,
                    const unsigned char* __restrict__ fend,
                    float* __restrict__ z0ws)
{
  const int b    = blockIdx.x;
  const int lane = threadIdx.x;

  // length = sum of prefix mask bytes
  const uint4* mrow = (const uint4*)(mask + (size_t)b * T_LEN);
  int s = 0;
  #pragma unroll
  for (int it = 0; it < T_LEN / (32 * 16); ++it) {
    const uint4 w = mrow[it * 32 + lane];
    s += byteSum(w.x) + byteSum(w.y) + byteSum(w.z) + byteSum(w.w);
  }
  #pragma unroll
  for (int d = 16; d; d >>= 1) s += __shfl_xor(s, d, 32);
  const int len = s;   // in [T/2, T]

  float acc = 0.0f;
  int carryTag = 0;
  for (int tc = 0; tc < T_LEN; tc += 32) {
    const int t = tc + lane;
    // one-hot row -> tag index (exactly one byte == 1)
    const uint4* trow = (const uint4*)(target + ((size_t)b * T_LEN + t) * NTAG);
    const uint4 w0 = trow[0], w1 = trow[1], w2 = trow[2], w3 = trow[3];
    unsigned wd[16] = {w0.x, w0.y, w0.z, w0.w, w1.x, w1.y, w1.z, w1.w,
                       w2.x, w2.y, w2.z, w2.w, w3.x, w3.y, w3.z, w3.w};
    int tag = -1;
    #pragma unroll
    for (int i = 0; i < 16; ++i) {
      const int cand = tagFromDword(wd[i], i * 4);
      tag = cand > tag ? cand : tag;
    }
    int prev = __shfl_up(tag, 1, 32);
    if (lane == 0) prev = carryTag;

    float contrib = 0.0f;
    if (t < len) {
      float base;
      if (t == 0) base = fstart[tag] ? IMPOSSIBLE_F : startt[tag];
      else        base = ftrans[prev * NTAG + tag] ? IMPOSSIBLE_F
                                                   : trans[prev * NTAG + tag];
      contrib = base + em[((size_t)b * T_LEN + t) * NTAG + tag];
      if (t == len - 1) contrib += fend[tag] ? IMPOSSIBLE_F : endt[tag];
    }
    #pragma unroll
    for (int d = 16; d; d >>= 1) contrib += __shfl_xor(contrib, d, 32);
    acc += contrib;
    carryTag = __shfl(tag, 31, 32);
  }
  if (lane == 0) z0ws[b] = acc;
}

// ---------------------------------------------------------------------------
// Unsupervised forward scan (base-2 log domain). One wave owns 16 chains.
// alpha layout == WMMA f32 C/D layout: reg v, lane l -> row m = v + 8*(l/16),
// tag j = 16*n + (l%16) for column-tile n in 0..3.
// ---------------------------------------------------------------------------
__global__ __launch_bounds__(32, 1)
void crf_unsup_scan(const float* __restrict__ em,
                    const unsigned char* __restrict__ mask,
                    const float* __restrict__ trans,
                    const float* __restrict__ startt,
                    const float* __restrict__ endt,
                    const unsigned char* __restrict__ ftrans,
                    const unsigned char* __restrict__ fstart,
                    const unsigned char* __restrict__ fend,
                    const float* __restrict__ z0ws,
                    float* __restrict__ out)
{
  __shared__ __align__(16) _Float16 pLDS[16][64];  // p transpose buffer
  __shared__ int lensLDS[16];

  const int lane = threadIdx.x;     // 0..31
  const int g    = lane >> 4;       // half-wave id
  const int lm   = lane & 15;
  const int cb   = blockIdx.x * 16; // chain (batch) base

  // ---- lengths for the 16 chains of this wave ----
  for (int cc = 0; cc < 16; ++cc) {
    const uint4* mrow = (const uint4*)(mask + (size_t)(cb + cc) * T_LEN);
    int s = 0;
    #pragma unroll
    for (int it = 0; it < T_LEN / (32 * 16); ++it) {
      const uint4 w = mrow[it * 32 + lane];
      s += byteSum(w.x) + byteSum(w.y) + byteSum(w.z) + byteSum(w.w);
    }
    #pragma unroll
    for (int d = 16; d; d >>= 1) s += __shfl_xor(s, d, 32);
    if (lane == 0) lensLDS[cc] = s;
  }
  __syncthreads();  // single wave: barrier -> S_NOP, but forces LDS waits
  int lenRow[8];
  #pragma unroll
  for (int v = 0; v < 8; ++v) lenRow[v] = lensLDS[v + 8 * g];

  int minLen = lenRow[0];
  #pragma unroll
  for (int v = 1; v < 8; ++v) minLen = lenRow[v] < minLen ? lenRow[v] : minLen;
  #pragma unroll
  for (int d = 16; d; d >>= 1) {
    const int o = __shfl_xor(minLen, d, 32);
    minLen = o < minLen ? o : minLen;
  }
  int tSplit = minLen - 1;
  if (tSplit < 1) tSplit = 1;

  // ---- resident E = 2^(trans_eff * log2e) in WMMA B layout ----
  // B as built here (contiguous K, same order as A): half h of lane l <-> K = 16*(l/16) + h, N = l%16.
  v16h Bm[2][4];
  #pragma unroll
  for (int k = 0; k < 2; ++k) {
    #pragma unroll
    for (int n = 0; n < 4; ++n) {
      v16h bt;
      #pragma unroll
      for (int h = 0; h < 16; ++h) {
        const int i = 32 * k + 16 * g + h;   // source tag (K)
        const int j = 16 * n + lm;           // dest tag (N)
        const float tv = ftrans[i * NTAG + j] ? IMPOSSIBLE_F
                                              : trans[i * NTAG + j];
        bt[h] = (_Float16)__builtin_amdgcn_exp2f(tv * L2E); // forbidden -> 0
      }
      Bm[k][n] = bt;
    }
  }

  float sEff2[4], eEff2[4];   // base-2 scaled
  #pragma unroll
  for (int n = 0; n < 4; ++n) {
    const int j = 16 * n + lm;
    sEff2[n] = (fstart[j] ? IMPOSSIBLE_F : startt[j]) * L2E;
    eEff2[n] = (fend[j]   ? IMPOSSIBLE_F : endt[j])   * L2E;
  }

  // per-lane emission row pointers (row m = v + 8*g, column lm folded in)
  const float* emRow[8];
  #pragma unroll
  for (int v = 0; v < 8; ++v)
    emRow[v] = em + (size_t)(cb + v + 8 * g) * T_LEN * NTAG + lm;

  // ---- alpha2_0 = (em[:,0,:] + start) * log2e ----
  float alpha[4][8], lastv[4][8], emv[4][8], emn[4][8];
  #pragma unroll
  for (int n = 0; n < 4; ++n)
    #pragma unroll
    for (int v = 0; v < 8; ++v) {
      const float a0 = emRow[v][16 * n] * L2E + sEff2[n];
      alpha[n][v] = a0;
      lastv[n][v] = a0;
    }
  // preload (scaled) emissions for t = 1
  #pragma unroll
  for (int n = 0; n < 4; ++n)
    #pragma unroll
    for (int v = 0; v < 8; ++v)
      emv[n][v] = emRow[v][(size_t)NTAG + 16 * n] * L2E;

  // ---- one scan step (base-2 log domain) ----
  auto step = [&](int t, bool doLatch) {
    // software pipeline: issue next step's emission loads first
    const int tn = (t + 1 < T_LEN) ? (t + 1) : t;
    #pragma unroll
    for (int n = 0; n < 4; ++n)
      #pragma unroll
      for (int v = 0; v < 8; ++v)
        emn[n][v] = emRow[v][(size_t)tn * NTAG + 16 * n] * L2E;

    // row max over 64 tags: across 4 tiles, then across 16 lanes
    float rmax[8];
    #pragma unroll
    for (int v = 0; v < 8; ++v) {
      float m = fmaxf(fmaxf(alpha[0][v], alpha[1][v]),
                      fmaxf(alpha[2][v], alpha[3][v]));
      m = fmaxf(m, __shfl_xor(m, 8, 32));
      m = fmaxf(m, __shfl_xor(m, 4, 32));
      m = fmaxf(m, __shfl_xor(m, 2, 32));
      m = fmaxf(m, __shfl_xor(m, 1, 32));
      rmax[v] = m;
    }

    // p = exp2(alpha2 - rmax2) -> LDS (D-layout -> row-major [m][i] halves)
    #pragma unroll
    for (int n = 0; n < 4; ++n)
      #pragma unroll
      for (int v = 0; v < 8; ++v)
        pLDS[v + 8 * g][16 * n + lm] =
            (_Float16)__builtin_amdgcn_exp2f(alpha[n][v] - rmax[v]);

    // pre-add (rmax2 + em2) while the LDS/WMMA pipe runs (independent)
    float pre[4][8];
    #pragma unroll
    for (int n = 0; n < 4; ++n)
      #pragma unroll
      for (int v = 0; v < 8; ++v)
        pre[n][v] = rmax[v] + emv[n][v];

    asm volatile("s_wait_dscnt 0x0" ::: "memory");  // LDS RAW fence (same wave)

    // A operand (16x32 f16): lane l holds row m = l%16;
    // half h -> K = 32k + 16*(l/16) + h (matches B).
    const v8h a00 = *(const v8h*)&pLDS[lm][     16 * g];
    const v8h a01 = *(const v8h*)&pLDS[lm][8 +  16 * g];
    const v8h a10 = *(const v8h*)&pLDS[lm][32 + 16 * g];
    const v8h a11 = *(const v8h*)&pLDS[lm][40 + 16 * g];
    const v16h A0 = __builtin_shufflevector(a00, a01, 0,1,2,3,4,5,6,7,8,9,10,11,12,13,14,15);
    const v16h A1 = __builtin_shufflevector(a10, a11, 0,1,2,3,4,5,6,7,8,9,10,11,12,13,14,15);

    // s = p @ E : 4 column tiles x 2 K-chunks = 8 WMMA
    #pragma unroll
    for (int n = 0; n < 4; ++n) {
      v8f c = {};
      c = __builtin_amdgcn_wmma_f32_16x16x32_f16(false, A0, false, Bm[0][n],
                                                 (short)0, c, false, false);
      c = __builtin_amdgcn_wmma_f32_16x16x32_f16(false, A1, false, Bm[1][n],
                                                 (short)0, c, false, false);
      #pragma unroll
      for (int v = 0; v < 8; ++v)
        alpha[n][v] = pre[n][v] + __builtin_amdgcn_logf(c[v]); // raw v_log_f32
    }

    if (doLatch) {
      #pragma unroll
      for (int v = 0; v < 8; ++v) {
        const bool hit = (t == lenRow[v] - 1);
        #pragma unroll
        for (int n = 0; n < 4; ++n)
          lastv[n][v] = hit ? alpha[n][v] : lastv[n][v];
      }
    }

    // rotate pipelined emissions
    #pragma unroll
    for (int n = 0; n < 4; ++n)
      #pragma unroll
      for (int v = 0; v < 8; ++v)
        emv[n][v] = emn[n][v];
  };

  // ---- sequential scan over T; no latch possible before minLen-1 ----
  for (int t = 1; t < tSplit; ++t) step(t, false);
  for (int t = tSplit; t < T_LEN; ++t) step(t, true);

  // ---- z1 = ln2 * log2sumexp2(last2 + end2), out = z1 - z0 ----
  #pragma unroll
  for (int n = 0; n < 4; ++n)
    #pragma unroll
    for (int v = 0; v < 8; ++v)
      lastv[n][v] += eEff2[n];

  #pragma unroll
  for (int v = 0; v < 8; ++v) {
    float m = fmaxf(fmaxf(lastv[0][v], lastv[1][v]),
                    fmaxf(lastv[2][v], lastv[3][v]));
    m = fmaxf(m, __shfl_xor(m, 8, 32));
    m = fmaxf(m, __shfl_xor(m, 4, 32));
    m = fmaxf(m, __shfl_xor(m, 2, 32));
    m = fmaxf(m, __shfl_xor(m, 1, 32));
    float ssum = __builtin_amdgcn_exp2f(lastv[0][v] - m) +
                 __builtin_amdgcn_exp2f(lastv[1][v] - m) +
                 __builtin_amdgcn_exp2f(lastv[2][v] - m) +
                 __builtin_amdgcn_exp2f(lastv[3][v] - m);
    ssum += __shfl_xor(ssum, 8, 32);
    ssum += __shfl_xor(ssum, 4, 32);
    ssum += __shfl_xor(ssum, 2, 32);
    ssum += __shfl_xor(ssum, 1, 32);
    const float z1 = LN2 * (m + __builtin_amdgcn_logf(ssum));
    if (lm == 0) {                 // lanes 0 and 16 -> rows v and v+8
      const int c = cb + v + 8 * g;
      out[c] = z1 - z0ws[c];       // -(z0 - z1)
    }
  }
}

// ---------------------------------------------------------------------------
extern "C" void kernel_launch(void* const* d_in, const int* in_sizes, int n_in,
                              void* d_out, int out_size, void* d_ws, size_t ws_size,
                              hipStream_t stream) {
  (void)in_sizes; (void)n_in; (void)out_size; (void)ws_size;
  const float*         emissions = (const float*)d_in[0];
  const unsigned char* mask      = (const unsigned char*)d_in[1];
  const unsigned char* target    = (const unsigned char*)d_in[2];
  const float*         trans     = (const float*)d_in[3];
  const float*         start_t   = (const float*)d_in[4];
  const float*         end_t     = (const float*)d_in[5];
  const unsigned char* ftrans    = (const unsigned char*)d_in[6];
  const unsigned char* fstart    = (const unsigned char*)d_in[7];
  const unsigned char* fend      = (const unsigned char*)d_in[8];
  float* out  = (float*)d_out;
  float* z0ws = (float*)d_ws;   // 64 floats of scratch

  crf_path_score<<<64, 32, 0, stream>>>(emissions, mask, target, trans,
                                        start_t, end_t, ftrans, fstart, fend,
                                        z0ws);
  crf_unsup_scan<<<4, 32, 0, stream>>>(emissions, mask, trans, start_t, end_t,
                                       ftrans, fstart, fend, z0ws, out);
}
